// ASPP_Adapter_77807627534483
// MI455X (gfx1250) — matrix operand order repaired and checked by reading, not compiled, not run on hardware
//
#include <hip/hip_runtime.h>
#include <stdint.h>

#define P_PIX 4096
#define EDIM 768
#define HDIM 64
#define WDIM 64
#define C2 384
#define KC_E (EDIM / 32)        // 24 K-chunks for K=768
#define NT_E (EDIM / 16)        // 48 N-tiles for N=768
#define TAP_STRIDE (KC_E * NT_E * 512)   // packed elems per conv tap = 589824

typedef __attribute__((ext_vector_type(16))) __bf16 v16bf;
typedef __attribute__((ext_vector_type(8)))  float  v8f;

union Frag16 {
  v16bf v;
  uint4 q[2];
  unsigned short s[16];
};

__device__ __forceinline__ unsigned short f2bf(float f) {
  unsigned int u = __float_as_uint(f);
  u += 0x7FFFu + ((u >> 16) & 1u);
  return (unsigned short)(u >> 16);
}

// ---------------- elementwise casts / packs ----------------

__global__ __launch_bounds__(256)
void cast_f32_bf16_kernel(const float* __restrict__ in, unsigned short* __restrict__ out, int n) {
  int i = blockIdx.x * 256 + threadIdx.x;
  if (i < n) out[i] = f2bf(in[i]);
}

// Pack a logical bf16 B matrix (K rows x N cols), element B[k][n] = src[k*sk + n*sn] (f32),
// into WMMA B-fragment layout:
//   dst[(((kc*NT) + nt)*32 + lane)*16 + j]  holds  B[kc*32 + (lane>>4)*16 + j][nt*16 + (lane&15)]
// Zero-filled for n >= N. total = (K/32)*NT*512.
__global__ __launch_bounds__(256)
void pack_b_frag_kernel(const float* __restrict__ src, int sk, int sn, int N,
                        unsigned short* __restrict__ dst, int NT, int total) {
  int i = blockIdx.x * 256 + threadIdx.x;
  if (i >= total) return;
  int j = i & 15;
  int lane = (i >> 4) & 31;
  int tile = i >> 9;
  int nt = tile % NT;
  int kc = tile / NT;
  int n = nt * 16 + (lane & 15);
  int k = kc * 32 + ((lane >> 4) << 4) + j;
  float v = (n < N) ? src[(size_t)k * sk + (size_t)n * sn] : 0.0f;
  dst[i] = f2bf(v);
}

// ---------------- generic bf16 WMMA GEMM (packed-fragment B) ----------------
// C[m,n] (+)= sum_k A[m*lda+k] * B[k][n]  (+ bias[n]); M from grid.x*16, N guarded on store.
__global__ __launch_bounds__(32)
void gemm_bf16_kernel(const unsigned short* __restrict__ A, int lda,
                      const unsigned short* __restrict__ Bpk, int NT,
                      float* __restrict__ C, int ldc,
                      int N, int Kdim, int accFlag, const float* __restrict__ bias) {
  int lane = threadIdx.x;
  int m0 = blockIdx.x * 16;
  int nt = blockIdx.y;
  int mlane = lane & 15;
  int col = nt * 16 + mlane;
  const unsigned short* ap = A + (size_t)(m0 + mlane) * lda + ((lane >> 4) << 4);
  const unsigned short* bp = Bpk + ((size_t)nt * 32 + lane) * 16;
  const size_t bstep = (size_t)NT * 512;
  int KC = Kdim >> 5;
  v8f acc = {};
  #pragma unroll 4
  for (int kc = 0; kc < KC; ++kc) {
    Frag16 a, b;
    a.q[0] = *(const uint4*)(ap);
    a.q[1] = *(const uint4*)(ap + 8);
    b.q[0] = *(const uint4*)(bp);
    b.q[1] = *(const uint4*)(bp + 8);
    __builtin_prefetch((const void*)(ap + 32), 0, 1);
    __builtin_prefetch((const void*)(bp + bstep), 0, 1);
    acc = __builtin_amdgcn_wmma_f32_16x16x32_bf16(false, a.v, false, b.v,
                                                  (short)0, acc, false, false);
    ap += 32;
    bp += bstep;
  }
  if (col < N) {
    int rbase = m0 + ((lane >> 4) << 3);
    float bia = bias ? bias[col] : 0.0f;
    #pragma unroll
    for (int v = 0; v < 8; ++v) {
      size_t off = (size_t)(rbase + v) * ldc + col;
      float val = acc[v] + bia;
      C[off] = accFlag ? (C[off] + val) : val;
    }
  }
}

// ---------------- conv3x3 as 9 shifted WMMA GEMMs (packed-fragment B) ----------------
// X bf16 [4096][768] (NHWC rows), Wpk packed per tap, out f32 [4096][768] (+bias)
__global__ __launch_bounds__(32)
void conv3x3_wmma_kernel(const unsigned short* __restrict__ X,
                         const unsigned short* __restrict__ Wpk,
                         const float* __restrict__ bias,
                         float* __restrict__ C) {
  int lane = threadIdx.x;
  int p0 = blockIdx.x * 16;
  int nt = blockIdx.y;
  int y = p0 >> 6;
  int x0 = p0 & 63;
  int mlane = lane & 15;
  int kHalf = (lane >> 4) << 4;
  int col = nt * 16 + mlane;
  const uint4 z4 = make_uint4(0u, 0u, 0u, 0u);
  v8f acc = {};
  for (int t = 0; t < 9; ++t) {
    int dy = t / 3 - 1, dx = t % 3 - 1;
    int ys = y + dy;
    if (ys < 0 || ys >= HDIM) continue;            // uniform over the wave
    int xs = x0 + mlane + dx;
    bool valid = (xs >= 0) && (xs < WDIM);
    int xc = min(max(xs, 0), WDIM - 1);
    const unsigned short* arow = X + (size_t)(ys * WDIM + xc) * EDIM + kHalf;
    const unsigned short* bp = Wpk + (size_t)t * TAP_STRIDE + ((size_t)nt * 32 + lane) * 16;
    #pragma unroll 2
    for (int kc = 0; kc < KC_E; ++kc) {
      Frag16 a, b;
      uint4 t0 = *(const uint4*)(arow);
      uint4 t1 = *(const uint4*)(arow + 8);
      a.q[0] = valid ? t0 : z4;
      a.q[1] = valid ? t1 : z4;
      b.q[0] = *(const uint4*)(bp);
      b.q[1] = *(const uint4*)(bp + 8);
      __builtin_prefetch((const void*)(arow + 32), 0, 1);
      acc = __builtin_amdgcn_wmma_f32_16x16x32_bf16(false, a.v, false, b.v,
                                                    (short)0, acc, false, false);
      arow += 32;
      bp += (size_t)NT_E * 512;
    }
  }
  int rbase = p0 + ((lane >> 4) << 3);
  float bia = bias[col];
  #pragma unroll
  for (int v = 0; v < 8; ++v)
    C[(size_t)(rbase + v) * EDIM + col] = acc[v] + bia;
}

// ---------------- offset/mask linears ----------------
__global__ __launch_bounds__(256)
void offmask_kernel(const float* __restrict__ om,
                    const float* __restrict__ offw, const float* __restrict__ offb,
                    const float* __restrict__ mskw, const float* __restrict__ mskb,
                    float* __restrict__ offs, float* __restrict__ mlog) {
  int idx = blockIdx.x * 256 + threadIdx.x;
  if (idx >= P_PIX * 27) return;
  int p = idx / 27, j = idx % 27;
  const float* row = om + (size_t)p * EDIM;
  const float* wrow;
  float acc;
  if (j < 18) { wrow = offw + (size_t)j * EDIM; acc = offb[j]; }
  else        { wrow = mskw + (size_t)(j - 18) * EDIM; acc = mskb[j - 18]; }
  for (int c = 0; c < EDIM; ++c) acc += row[c] * wrow[c];
  if (j < 18) offs[(size_t)p * 18 + j] = acc;
  else        mlog[(size_t)p * 9 + (j - 18)] = acc;
}

__global__ __launch_bounds__(256)
void softmax9_kernel(float* __restrict__ m) {
  int p = blockIdx.x * 256 + threadIdx.x;
  if (p >= P_PIX) return;
  float* r = m + (size_t)p * 9;
  float mx = r[0];
  for (int j = 1; j < 9; ++j) mx = fmaxf(mx, r[j]);
  float e[9], s = 0.f;
  for (int j = 0; j < 9; ++j) { e[j] = __expf(r[j] - mx); s += e[j]; }
  for (int j = 0; j < 9; ++j) r[j] = e[j] / s;
}

// ---------------- bilinear deform sampling (one tap) ----------------
// xin f32 NHWC [4096][768]; mask_used[t][p] = maskf[t*4096+p] (torch raw-reshape quirk)
__global__ __launch_bounds__(256)
void deform_samp_kernel(const float* __restrict__ xin,
                        const float* __restrict__ offs,
                        const float* __restrict__ maskf,
                        int t, unsigned short* __restrict__ samp) {
  int p = blockIdx.x;
  int y = p >> 6, x = p & 63;
  int ky = t / 3, kx = t % 3;
  float dy = offs[(size_t)p * 18 + 2 * t];
  float dx = offs[(size_t)p * 18 + 2 * t + 1];
  float py = (float)y - 1.0f + (float)ky + dy;
  float px = (float)x - 1.0f + (float)kx + dx;
  float y0f = floorf(py), x0f = floorf(px);
  float wy1 = py - y0f, wx1 = px - x0f;
  float wy0 = 1.f - wy1, wx0 = 1.f - wx1;
  int y0 = (int)y0f, x0i = (int)x0f;
  float m = maskf[(size_t)t * P_PIX + p];

  const float* ptr[4];
  float w[4];
  int yy[4] = { y0, y0, y0 + 1, y0 + 1 };
  int xx[4] = { x0i, x0i + 1, x0i, x0i + 1 };
  float ww[4] = { wy0 * wx0, wy0 * wx1, wy1 * wx0, wy1 * wx1 };
  #pragma unroll
  for (int k = 0; k < 4; ++k) {
    bool v = (yy[k] >= 0) && (yy[k] < HDIM) && (xx[k] >= 0) && (xx[k] < WDIM);
    int yc = min(max(yy[k], 0), HDIM - 1);
    int xc = min(max(xx[k], 0), WDIM - 1);
    ptr[k] = xin + (size_t)(yc * WDIM + xc) * EDIM;
    w[k] = v ? ww[k] : 0.f;
  }
  for (int c = threadIdx.x; c < EDIM; c += 256) {
    float v = w[0] * ptr[0][c] + w[1] * ptr[1][c] + w[2] * ptr[2][c] + w[3] * ptr[3][c];
    samp[(size_t)p * EDIM + c] = f2bf(m * v);
  }
}

// ---------------- BN (train statistics) ----------------
__global__ __launch_bounds__(256)
void bn_stats_kernel(const float* __restrict__ d, float* __restrict__ stats) {
  __shared__ float ssum[256], ssq[256];
  int c = blockIdx.x;
  float s = 0.f, q = 0.f;
  for (int p = threadIdx.x; p < P_PIX; p += 256) {
    float v = d[(size_t)p * EDIM + c];
    s += v; q += v * v;
  }
  ssum[threadIdx.x] = s; ssq[threadIdx.x] = q;
  __syncthreads();
  for (int st = 128; st > 0; st >>= 1) {
    if (threadIdx.x < st) { ssum[threadIdx.x] += ssum[threadIdx.x + st]; ssq[threadIdx.x] += ssq[threadIdx.x + st]; }
    __syncthreads();
  }
  if (threadIdx.x == 0) {
    float mu = ssum[0] / (float)P_PIX;
    stats[c * 2] = mu;
    stats[c * 2 + 1] = ssq[0] / (float)P_PIX - mu * mu;
  }
}

__global__ __launch_bounds__(256)
void bn_relu_cast_kernel(const float* __restrict__ d, const float* __restrict__ stats,
                         const float* __restrict__ g, const float* __restrict__ b,
                         unsigned short* __restrict__ out) {
  int i = blockIdx.x * 256 + threadIdx.x;
  if (i >= P_PIX * EDIM) return;
  int c = i % EDIM;
  float v = (d[i] - stats[c * 2]) * rsqrtf(stats[c * 2 + 1] + 1e-5f) * g[c] + b[c];
  out[i] = f2bf(fmaxf(v, 0.f));
}

// ---------------- small reductions / GEMV ----------------
__global__ __launch_bounds__(256)
void rowdot_kernel(const float* __restrict__ A, int ld, const float* __restrict__ v,
                   int n, float* __restrict__ out, int rows, int mode) {
  int r = blockIdx.x * 256 + threadIdx.x;
  if (r >= rows) return;
  const float* a = A + (size_t)r * ld;
  float s = 0.f;
  for (int i = 0; i < n; ++i) s += a[i] * v[i];
  out[r] = mode ? (1.f / (1.f + __expf(-s))) : s;
}

__global__ __launch_bounds__(256)
void softmax_vec_kernel(float* __restrict__ v, int n) {
  __shared__ float sh[256];
  float mx = -1e30f;
  for (int i = threadIdx.x; i < n; i += 256) mx = fmaxf(mx, v[i]);
  sh[threadIdx.x] = mx; __syncthreads();
  for (int st = 128; st > 0; st >>= 1) {
    if (threadIdx.x < st) sh[threadIdx.x] = fmaxf(sh[threadIdx.x], sh[threadIdx.x + st]);
    __syncthreads();
  }
  mx = sh[0]; __syncthreads();
  float s = 0.f;
  for (int i = threadIdx.x; i < n; i += 256) { float e = __expf(v[i] - mx); v[i] = e; s += e; }
  sh[threadIdx.x] = s; __syncthreads();
  for (int st = 128; st > 0; st >>= 1) {
    if (threadIdx.x < st) sh[threadIdx.x] += sh[threadIdx.x + st];
    __syncthreads();
  }
  s = sh[0];
  for (int i = threadIdx.x; i < n; i += 256) v[i] /= s;
}

__global__ __launch_bounds__(256)
void colreduce_dot_kernel(const float* __restrict__ mat, const float* __restrict__ vec,
                          float* __restrict__ out, int rows, int cols) {
  __shared__ float sh[256];
  int o = blockIdx.x;
  float s = 0.f;
  for (int p = threadIdx.x; p < rows; p += 256) s += mat[(size_t)p * cols + o] * vec[p];
  sh[threadIdx.x] = s; __syncthreads();
  for (int st = 128; st > 0; st >>= 1) {
    if (threadIdx.x < st) sh[threadIdx.x] += sh[threadIdx.x + st];
    __syncthreads();
  }
  if (threadIdx.x == 0) out[o] = sh[0];
}

__global__ __launch_bounds__(256)
void colreduce_max_kernel(const float* __restrict__ mat, float* __restrict__ out,
                          int rows, int cols) {
  __shared__ float sh[256];
  int o = blockIdx.x;
  float s = -1e30f;
  for (int p = threadIdx.x; p < rows; p += 256) s = fmaxf(s, mat[(size_t)p * cols + o]);
  sh[threadIdx.x] = s; __syncthreads();
  for (int st = 128; st > 0; st >>= 1) {
    if (threadIdx.x < st) sh[threadIdx.x] = fmaxf(sh[threadIdx.x], sh[threadIdx.x + st]);
    __syncthreads();
  }
  if (threadIdx.x == 0) out[o] = sh[0];
}

__global__ __launch_bounds__(256)
void ln_sigmoid_kernel(const float* __restrict__ wz, const float* __restrict__ g,
                       const float* __restrict__ b, float* __restrict__ sig) {
  __shared__ float sh[256], sh2[256];
  float s = 0.f, q = 0.f;
  for (int i = threadIdx.x; i < EDIM; i += 256) { float v = wz[i]; s += v; q += v * v; }
  sh[threadIdx.x] = s; sh2[threadIdx.x] = q; __syncthreads();
  for (int st = 128; st > 0; st >>= 1) {
    if (threadIdx.x < st) { sh[threadIdx.x] += sh[threadIdx.x + st]; sh2[threadIdx.x] += sh2[threadIdx.x + st]; }
    __syncthreads();
  }
  float mu = sh[0] / (float)EDIM;
  float var = sh2[0] / (float)EDIM - mu * mu;
  float inv = rsqrtf(var + 1e-5f);
  for (int i = threadIdx.x; i < EDIM; i += 256) {
    float v = (wz[i] - mu) * inv * g[i] + b[i];
    sig[i] = 1.f / (1.f + __expf(-v));
  }
}

__global__ __launch_bounds__(256)
void x2_kernel(const float* __restrict__ cat, const float* __restrict__ sig,
               float* __restrict__ x2, unsigned short* __restrict__ x2bf) {
  int i = blockIdx.x * 256 + threadIdx.x;
  if (i >= P_PIX * EDIM) return;
  float v = sig[i % EDIM] * cat[i];
  x2[i] = v;
  x2bf[i] = f2bf(v);
}

__global__ __launch_bounds__(256)
void final_ln_kernel(const float* __restrict__ x2, const float* __restrict__ cat,
                     const float* __restrict__ x5, const float* __restrict__ wzp,
                     const float* __restrict__ g, const float* __restrict__ b,
                     float* __restrict__ out) {
  __shared__ float row[EDIM];
  __shared__ float sh[256], sh2[256];
  int p = blockIdx.x;
  float wp = wzp[p];
  float s = 0.f, q = 0.f;
  for (int c = threadIdx.x; c < EDIM; c += 256) {
    float v = wp * x2[(size_t)p * EDIM + c] + cat[(size_t)p * EDIM + c] + x5[(size_t)p * EDIM + c];
    row[c] = v; s += v; q += v * v;
  }
  sh[threadIdx.x] = s; sh2[threadIdx.x] = q; __syncthreads();
  for (int st = 128; st > 0; st >>= 1) {
    if (threadIdx.x < st) { sh[threadIdx.x] += sh[threadIdx.x + st]; sh2[threadIdx.x] += sh2[threadIdx.x + st]; }
    __syncthreads();
  }
  float mu = sh[0] / (float)EDIM;
  float var = sh2[0] / (float)EDIM - mu * mu;
  float inv = rsqrtf(var + 1e-5f);
  for (int c = threadIdx.x; c < EDIM; c += 256)
    out[(size_t)p * EDIM + c] = (row[c] - mu) * inv * g[c] + b[c];
}

// ---------------- host ----------------
extern "C" void kernel_launch(void* const* d_in, const int* in_sizes, int n_in,
                              void* d_out, int out_size, void* d_ws, size_t ws_size,
                              hipStream_t stream) {
  (void)in_sizes; (void)n_in; (void)out_size; (void)ws_size;
  const float* xin[5];
  for (int i = 0; i < 5; ++i) xin[i] = (const float*)d_in[i];
  const float* conv_w = (const float*)d_in[5];
  const float* conv_b = (const float*)d_in[6];
  const float* off_w  = (const float*)d_in[7];
  const float* off_b  = (const float*)d_in[8];
  const float* msk_w  = (const float*)d_in[9];
  const float* msk_b  = (const float*)d_in[10];
  const float* bn_g   = (const float*)d_in[11];
  const float* bn_b   = (const float*)d_in[12];
  const float* chq_w  = (const float*)d_in[13];
  const float* chz_w  = (const float*)d_in[15];
  const float* ln_g   = (const float*)d_in[16];
  const float* ln_b   = (const float*)d_in[17];
  const float* norm_g = (const float*)d_in[28];
  const float* norm_b = (const float*)d_in[29];
  float* out = (float*)d_out;

  char* base = (char*)d_ws;
  size_t cur = 0;
  auto carve = [&](size_t bytes) -> void* {
    void* p = base + cur;
    cur = (cur + bytes + 255) & ~(size_t)255;
    return p;
  };
  // packed conv weights: 9 taps, each KC_E*NT_E*512 bf16 elems
  unsigned short* wpk = (unsigned short*)carve((size_t)9 * TAP_STRIDE * 2);
  unsigned short* xbf[5];
  for (int i = 0; i < 5; ++i) xbf[i] = (unsigned short*)carve((size_t)P_PIX * EDIM * 2);
  float* om    = (float*)carve((size_t)P_PIX * EDIM * 4);
  float* offs  = (float*)carve((size_t)P_PIX * 18 * 4);
  float* maskb = (float*)carve((size_t)P_PIX * 9 * 4);
  unsigned short* samp = (unsigned short*)carve((size_t)P_PIX * EDIM * 2);
  float* dconv = (float*)carve((size_t)P_PIX * EDIM * 4);
  float* stats = (float*)carve((size_t)EDIM * 2 * 4);
  unsigned short* blkbf = (unsigned short*)carve((size_t)P_PIX * EDIM * 2);
  float* cat   = (float*)carve((size_t)P_PIX * EDIM * 4);
  unsigned short* catbf = (unsigned short*)carve((size_t)P_PIX * EDIM * 2);
  const int CH[5] = {30, 100, 150, 220, 268};
  const int chOf[8] = {30, 100, 150, 150, 220, 220, 268, 268};
  unsigned short* pwpk[8];
  int pwNT[8];
  for (int j = 0; j < 8; ++j) {
    pwNT[j] = (chOf[j] + 15) / 16;
    pwpk[j] = (unsigned short*)carve((size_t)KC_E * pwNT[j] * 512 * 2);
  }
  unsigned short* chvpk = (unsigned short*)carve((size_t)KC_E * (C2 / 16) * 512 * 2);
  unsigned short* spqpk = (unsigned short*)carve((size_t)KC_E * (C2 / 16) * 512 * 2);
  unsigned short* spvpk = (unsigned short*)carve((size_t)KC_E * (C2 / 16) * 512 * 2);
  float* qp     = (float*)carve((size_t)P_PIX * 4);
  float* buf_po = (float*)carve((size_t)P_PIX * C2 * 4);
  float* wvq    = (float*)carve((size_t)C2 * 4);
  float* wzc    = (float*)carve((size_t)EDIM * 4);
  float* sig    = (float*)carve((size_t)EDIM * 4);
  float* x2     = (float*)carve((size_t)P_PIX * EDIM * 4);
  unsigned short* x2bf = (unsigned short*)carve((size_t)P_PIX * EDIM * 2);
  float* q384   = (float*)carve((size_t)C2 * 4);
  float* wzp    = (float*)carve((size_t)P_PIX * 4);

  auto castN = [&](const float* src, unsigned short* dst, int n) {
    cast_f32_bf16_kernel<<<(n + 255) / 256, 256, 0, stream>>>(src, dst, n);
  };
  auto packB = [&](const float* src, int sk, int sn, int N, unsigned short* dst, int NT) {
    int total = KC_E * NT * 512;
    pack_b_frag_kernel<<<(total + 255) / 256, 256, 0, stream>>>(src, sk, sn, N, dst, NT, total);
  };
  auto gemm = [&](const unsigned short* A, int lda, const unsigned short* Bpk, int NT,
                  float* C, int ldc, int M, int N, int K, int acc, const float* bias) {
    dim3 g(M / 16, NT);
    gemm_bf16_kernel<<<g, 32, 0, stream>>>(A, lda, Bpk, NT, C, ldc, N, K, acc, bias);
  };

  // ---- precision packing ----
  // conv taps: B[k=c][n=o] = conv_w[(o*768+c)*9 + t]  ->  src=conv_w+t, sk=9, sn=768*9
  for (int t = 0; t < 9; ++t)
    packB(conv_w + t, 9, EDIM * 9, EDIM, wpk + (size_t)t * TAP_STRIDE, NT_E);
  for (int i = 0; i < 5; ++i) castN(xin[i], xbf[i], P_PIX * EDIM);
  // 1x1 projection weights w[CH][768]: B[k][n] = w[n*768+k]
  for (int j = 0; j < 8; ++j)
    packB((const float*)d_in[20 + j], 1, EDIM, chOf[j], pwpk[j], pwNT[j]);
  packB((const float*)d_in[14], 1, EDIM, C2, chvpk, C2 / 16);  // chv_w
  packB((const float*)d_in[18], 1, EDIM, C2, spqpk, C2 / 16);  // spq_w
  packB((const float*)d_in[19], 1, EDIM, C2, spvpk, C2 / 16);  // spv_w

  // ---- 5 deformable-conv blocks + 1x1 projections into concat buffer ----
  const int coff[5] = {0, 30, 130, 280, 500};
  const unsigned short* pwA[5] = {pwpk[0], pwpk[1], pwpk[2], pwpk[4], pwpk[6]};
  const unsigned short* pwB[5] = {nullptr, nullptr, pwpk[3], pwpk[5], pwpk[7]};
  const int ntA[5] = {pwNT[0], pwNT[1], pwNT[2], pwNT[4], pwNT[6]};
  for (int i = 0; i < 5; ++i) {
    conv3x3_wmma_kernel<<<dim3(P_PIX / 16, NT_E), 32, 0, stream>>>(xbf[i], wpk, conv_b, om);
    offmask_kernel<<<(P_PIX * 27 + 255) / 256, 256, 0, stream>>>(om, off_w, off_b, msk_w, msk_b, offs, maskb);
    softmax9_kernel<<<(P_PIX + 255) / 256, 256, 0, stream>>>(maskb);
    for (int t = 0; t < 9; ++t) {
      deform_samp_kernel<<<P_PIX, 256, 0, stream>>>(xin[i], offs, maskb, t, samp);
      // bias omitted: removed exactly by the following BatchNorm
      gemm(samp, EDIM, wpk + (size_t)t * TAP_STRIDE, NT_E,
           dconv, EDIM, P_PIX, EDIM, EDIM, t > 0 ? 1 : 0, nullptr);
    }
    bn_stats_kernel<<<EDIM, 256, 0, stream>>>(dconv, stats);
    bn_relu_cast_kernel<<<(P_PIX * EDIM + 255) / 256, 256, 0, stream>>>(dconv, stats, bn_g, bn_b, blkbf);
    gemm(blkbf, EDIM, pwA[i], ntA[i], cat + coff[i], EDIM, P_PIX, CH[i], EDIM, 0, nullptr);
    if (pwB[i])
      gemm(xbf[i], EDIM, pwB[i], ntA[i], cat + coff[i], EDIM, P_PIX, CH[i], EDIM, 1, nullptr);
  }

  // ---- CAM: channel attention ----
  castN(cat, catbf, P_PIX * EDIM);
  rowdot_kernel<<<(P_PIX + 255) / 256, 256, 0, stream>>>(cat, EDIM, chq_w, EDIM, qp, P_PIX, 0);
  softmax_vec_kernel<<<1, 256, 0, stream>>>(qp, P_PIX);
  gemm(catbf, EDIM, chvpk, C2 / 16, buf_po, C2, P_PIX, C2, EDIM, 0, nullptr);
  colreduce_dot_kernel<<<C2, 256, 0, stream>>>(buf_po, qp, wvq, P_PIX, C2);
  rowdot_kernel<<<(EDIM + 255) / 256, 256, 0, stream>>>(chz_w, C2, wvq, C2, wzc, EDIM, 0);
  ln_sigmoid_kernel<<<1, 256, 0, stream>>>(wzc, ln_g, ln_b, sig);
  x2_kernel<<<(P_PIX * EDIM + 255) / 256, 256, 0, stream>>>(cat, sig, x2, x2bf);

  // ---- CAM: spatial attention ----
  gemm(x2bf, EDIM, spqpk, C2 / 16, buf_po, C2, P_PIX, C2, EDIM, 0, nullptr);
  colreduce_max_kernel<<<C2, 256, 0, stream>>>(buf_po, q384, P_PIX, C2);
  softmax_vec_kernel<<<1, 256, 0, stream>>>(q384, C2);
  gemm(x2bf, EDIM, spvpk, C2 / 16, buf_po, C2, P_PIX, C2, EDIM, 0, nullptr);
  rowdot_kernel<<<(P_PIX + 255) / 256, 256, 0, stream>>>(buf_po, C2, q384, C2, wzp, P_PIX, 1);

  // ---- residual + final LayerNorm ----
  final_ln_kernel<<<P_PIX, 256, 0, stream>>>(x2, cat, xin[4], wzp, norm_g, norm_b, out);
}
